// ConvLstm_79035988181348
// MI455X (gfx1250) — matrix operand order repaired and checked by reading, not compiled, not run on hardware
//
#include <hip/hip_runtime.h>
#include <math.h>

// ---------------- problem constants ----------------
#define HW    1024          // 32*32
#define WW    32
#define NB    64            // batch
#define NSTEP 8             // timesteps
#define NCOL  65536         // NB*HW  (GEMM N for conv layers)

#define KCMAX 256           // K-chunk staged in LDS per block
#define LDSPAD 8            // bf16 pad per LDS row (4 DWORDs, from TDM pad)

typedef float  v8f   __attribute__((ext_vector_type(8)));
typedef __bf16 v16bf __attribute__((ext_vector_type(16)));
typedef __bf16 v8bf  __attribute__((ext_vector_type(8)));
typedef unsigned int u32x4 __attribute__((ext_vector_type(4)));
typedef int          i32x8 __attribute__((ext_vector_type(8)));
typedef int          i32x4 __attribute__((ext_vector_type(4)));

// =====================================================================
// TDM: load a [32 x KC] bf16 tile of A (row stride K elements) into LDS,
// padding each row by 4 DWORDs so LDS row stride = KC + 8 bf16.
// D# per CDNA5 ISA 8.3/8.4: group0 {count,lds_addr,global_addr,type=2},
// group1 {data_size=2B, pad_enable, pad_interval=log2(KC)-2, pad_amount=4dw,
//         tensor_dim0=K, tensor_dim1=32, tile_dim0=KC, tile_dim1=32,
//         tensor_dim0_stride=K}.  Groups 2/3 zero (2-D tensor).
// Toolchain here exposes the 6-arg builtin (clang-23 / therock-10.0 form).
// =====================================================================
__device__ inline void tdm_load_a_tile(const __bf16* gsrc, unsigned ldsOff,
                                       int K, int KC)
{
    unsigned long long ga = (unsigned long long)(uintptr_t)gsrc;
    u32x4 g0;
    g0.x = 1u;                                            // count = 1
    g0.y = ldsOff;                                        // lds_addr (bytes)
    g0.z = (unsigned)(ga & 0xffffffffull);                // global_addr lo
    g0.w = (unsigned)((ga >> 32) & 0x1ffffffull)          // global_addr hi
         | (2u << 30);                                    // type = 2 (image)

    int lg = 31 - __builtin_clz((unsigned)KC);            // log2(KC)
    int interval = lg - 2;                                // 2*2^i dwords = KC/2 dw
    i32x8 g1;
    g1[0] = (1 << 16)                                     // data_size = 2 bytes
          | (1 << 20)                                     // pad_enable
          | (interval << 22)                              // pad_interval
          | (3 << 25);                                    // pad_amount = 4 dwords
    g1[1] = (K & 0xffff) << 16;                           // tensor_dim0 [15:0]
    g1[2] = ((K >> 16) & 0xffff) | (32 << 16);            // dim0 hi | tensor_dim1=32
    g1[3] = (KC << 16);                                   // dim1 hi(0) | tile_dim0
    g1[4] = 32;                                           // tile_dim1=32, tile_dim2=0
    g1[5] = K;                                            // tensor_dim0_stride lo32
    g1[6] = 0;                                            // stride0 hi | stride1 lo
    g1[7] = 0;
    i32x4 g2 = {0, 0, 0, 0};
    i32x4 g3 = {0, 0, 0, 0};
    i32x8 g4 = {0, 0, 0, 0, 0, 0, 0, 0};
    __builtin_amdgcn_tensor_load_to_lds(g0, g1, g2, g3, g4, 0);
}

// =====================================================================
// WMMA GEMM:  C[M,N] (f32) = A[M,K] (bf16, row-major) * B[N,K]^T (bf16)
// K mult of 32 (and of min(K,256)), M mult of 32, N mult of 64.
// Block: 256 thr (8 waves), tile M=32 x N=512.  Wave: 32x64 strip,
// 2x4 accumulators -> 8 v_wmma per 32-wide K-step.  A staged in LDS by TDM.
// A 16x32 / B 32x16 both use contiguous k = 16*(lane/16)+s; f32 C/D layout per CDNA5 ISA 7.12.2.
// =====================================================================
__global__ __launch_bounds__(256)
void gemm_wmma_bf16(const __bf16* __restrict__ A, const __bf16* __restrict__ Bm,
                    float* __restrict__ C, const float* __restrict__ bias,
                    int M, int N, int K, int accFlag, int reluFlag)
{
    __shared__ __align__(16) __bf16 As[32 * (KCMAX + LDSPAD)];

    const int lane = threadIdx.x & 31;
    const int wave = __builtin_amdgcn_readfirstlane(threadIdx.x >> 5); // scalar
    const int m0   = blockIdx.y * 32;
    const int n0   = (blockIdx.x * 8 + wave) * 64;
    const int KC   = K < KCMAX ? K : KCMAX;
    const int ldsk = KC + LDSPAD;                 // LDS row stride (bf16)
    const int g    = lane >> 4;                   // half-wave selector
    const int l16  = lane & 15;
    const bool active = (n0 < N);                 // scalar (wave from sgpr)

    v8f acc[2][4];
#pragma unroll
    for (int mt = 0; mt < 2; ++mt)
#pragma unroll
        for (int j = 0; j < 4; ++j)
#pragma unroll
            for (int e = 0; e < 8; ++e) acc[mt][j][e] = 0.0f;

    for (int k0 = 0; k0 < K; k0 += KC) {
        if (wave == 0) {
            tdm_load_a_tile(A + (size_t)m0 * K + k0,
                            (unsigned)(uintptr_t)As, K, KC);
            __builtin_amdgcn_s_wait_tensorcnt(0);
        }
        __syncthreads();

        if (active) {
            for (int kk = 0; kk < KC; kk += 32) {
                // A fragments from LDS: lane m = mt*16 + l16,
                // V0-3 <- K=kk+g*16..+7, V4-7 <- K=kk+g*16+8..+15
                v16bf af[2];
#pragma unroll
                for (int mt = 0; mt < 2; ++mt) {
                    const __bf16* ap = As + (size_t)(mt * 16 + l16) * ldsk + kk;
                    v8bf alo = *(const v8bf*)(ap + g * 16);
                    v8bf ahi = *(const v8bf*)(ap + g * 16 + 8);
#pragma unroll
                    for (int e = 0; e < 8; ++e) { af[mt][e] = alo[e]; af[mt][e + 8] = ahi[e]; }
                }
#pragma unroll
                for (int j = 0; j < 4; ++j) {
                    // B: lane holds 16 contiguous K at k0+kk+g*16, col n = l16
                    const int n = n0 + j * 16 + l16;
                    v16bf bfrag = *(const v16bf*)(Bm + (size_t)n * K + k0 + kk + g * 16);
                    acc[0][j] = __builtin_amdgcn_wmma_f32_16x16x32_bf16(
                        false, af[0], false, bfrag, (short)0, acc[0][j], false, false);
                    acc[1][j] = __builtin_amdgcn_wmma_f32_16x16x32_bf16(
                        false, af[1], false, bfrag, (short)0, acc[1][j], false, false);
                }
            }
        }
        __syncthreads();
    }

    if (active) {
        // C/D layout: VGPR r -> M = r (lanes 0-15) / r+8 (lanes 16-31)
#pragma unroll
        for (int mt = 0; mt < 2; ++mt)
#pragma unroll
            for (int j = 0; j < 4; ++j) {
                const int n = n0 + j * 16 + l16;
#pragma unroll
                for (int r = 0; r < 8; ++r) {
                    const int m = m0 + mt * 16 + r + g * 8;
                    float v = acc[mt][j][r];
                    if (bias)     v += bias[m];
                    const size_t off = (size_t)m * N + n;
                    if (accFlag)  v += C[off];
                    if (reluFlag) v  = fmaxf(v, 0.0f);
                    C[off] = v;
                }
            }
    }
}

// =====================================================================
// im2row for 2x2 "same"(pad lo=0, hi=1) conv: src f32 [B, C, 32, 32]
// -> col bf16 [NCOL, Kpad], row n = (b*1024 + i*32 + j),
// k = c*4 + di*2 + dj  (matches [O,I,2,2] weight flattening). k>=C*4 -> 0.
// =====================================================================
__global__ void im2row_kernel(const float* __restrict__ src, __bf16* __restrict__ col,
                              int C, int Kpad, int batchStride, int srcOff)
{
    size_t idx = (size_t)blockIdx.x * blockDim.x + threadIdx.x;
    size_t total = (size_t)NCOL * Kpad;
    if (idx >= total) return;
    int k = (int)(idx % Kpad);
    int n = (int)(idx / Kpad);
    float v = 0.0f;
    if (k < C * 4) {
        int c = k >> 2, di = (k >> 1) & 1, dj = k & 1;
        int b = n >> 10, p = n & 1023;
        int i = p >> 5,  j = p & 31;
        int ii = i + di, jj = j + dj;
        if (ii < WW && jj < WW)
            v = src[(size_t)b * batchStride + srcOff + (size_t)c * HW + ii * WW + jj];
    }
    col[idx] = (__bf16)v;
}

// hard_sigmoid(i,f,o), tanh(g), c' = f*c + i*tanh(g), h = o*tanh(c')
// Z f32 [4F, NCOL] gate-major (Keras i,f,g,o blocks); h,c f32 [B,F,HW]
__global__ void lstm_gates_kernel(const float* __restrict__ Z, float* __restrict__ h,
                                  float* __restrict__ c, int F)
{
    size_t idx = (size_t)blockIdx.x * blockDim.x + threadIdx.x;
    size_t total = (size_t)NB * F * HW;
    if (idx >= total) return;
    int p  = (int)(idx % HW);
    int ch = (int)((idx / HW) % F);
    int b  = (int)(idx / ((size_t)F * HW));
    size_t n = (size_t)b * HW + p;
    float zi = Z[(size_t)(0 * F + ch) * NCOL + n];
    float zf = Z[(size_t)(1 * F + ch) * NCOL + n];
    float zg = Z[(size_t)(2 * F + ch) * NCOL + n];
    float zo = Z[(size_t)(3 * F + ch) * NCOL + n];
    float ig = fminf(fmaxf(0.2f * zi + 0.5f, 0.f), 1.f);
    float fg = fminf(fmaxf(0.2f * zf + 0.5f, 0.f), 1.f);
    float og = fminf(fmaxf(0.2f * zo + 0.5f, 0.f), 1.f);
    float gg = tanhf(zg);
    float cn = fg * c[idx] + ig * gg;
    c[idx] = cn;
    h[idx] = og * tanhf(cn);
}

// conv weight f32 [M, C,2,2] -> bf16 [M, Kpad] (zero-pad k >= C*4)
__global__ void cvt_w_kernel(const float* __restrict__ src, __bf16* __restrict__ dst,
                             int M, int K, int Kpad)
{
    size_t idx = (size_t)blockIdx.x * blockDim.x + threadIdx.x;
    if (idx >= (size_t)M * Kpad) return;
    int m = (int)(idx / Kpad), k = (int)(idx % Kpad);
    dst[idx] = (k < K) ? (__bf16)src[(size_t)m * K + k] : (__bf16)0.0f;
}

// f32 [rows, cols] -> bf16 [cols, rows]   (transpose + convert)
__global__ void cvt_t_kernel(const float* __restrict__ src, __bf16* __restrict__ dst,
                             int rows, int cols)
{
    size_t idx = (size_t)blockIdx.x * blockDim.x + threadIdx.x;
    if (idx >= (size_t)rows * cols) return;
    int r = (int)(idx / cols), c = (int)(idx % cols);
    dst[(size_t)c * rows + r] = (__bf16)src[idx];
}

__global__ void cvt_flat_kernel(const float* __restrict__ src, __bf16* __restrict__ dst,
                                size_t n)
{
    size_t idx = (size_t)blockIdx.x * blockDim.x + threadIdx.x;
    if (idx < n) dst[idx] = (__bf16)src[idx];
}

__global__ void zero_kernel(float* __restrict__ p, size_t n)
{
    size_t idx = (size_t)blockIdx.x * blockDim.x + threadIdx.x;
    if (idx < n) p[idx] = 0.0f;
}

// final 1024 -> 2 GEMV + softmax.  a5T f32 [1024, 64] (channel-major)
__global__ void head_kernel(const float* __restrict__ a5T, const float* __restrict__ W6,
                            const float* __restrict__ b6, float* __restrict__ out)
{
    int b = threadIdx.x;
    if (b >= NB) return;
    float s0 = b6[0], s1 = b6[1];
    for (int k = 0; k < 1024; ++k) {
        float a = a5T[(size_t)k * NB + b];
        s0 += a * W6[2 * k];
        s1 += a * W6[2 * k + 1];
    }
    float mx = fmaxf(s0, s1);
    float e0 = __expf(s0 - mx), e1 = __expf(s1 - mx);
    float inv = 1.0f / (e0 + e1);
    out[2 * b]     = e0 * inv;
    out[2 * b + 1] = e1 * inv;
}

// =====================================================================
static inline unsigned cdiv(size_t a, size_t b) { return (unsigned)((a + b - 1) / b); }

extern "C" void kernel_launch(void* const* d_in, const int* in_sizes, int n_in,
                              void* d_out, int out_size, void* d_ws, size_t ws_size,
                              hipStream_t stream)
{
    const float* x   = (const float*)d_in[0];
    const float* Wx1 = (const float*)d_in[1];
    const float* Wh1 = (const float*)d_in[2];
    const float* b1  = (const float*)d_in[3];
    const float* Wx2 = (const float*)d_in[4];
    const float* Wh2 = (const float*)d_in[5];
    const float* b2  = (const float*)d_in[6];
    const float* Wx3 = (const float*)d_in[7];
    const float* Wh3 = (const float*)d_in[8];
    const float* b3  = (const float*)d_in[9];
    const float* W4  = (const float*)d_in[10];
    const float* b4  = (const float*)d_in[11];
    const float* W5  = (const float*)d_in[12];
    const float* b5  = (const float*)d_in[13];
    const float* W6  = (const float*)d_in[14];
    const float* b6  = (const float*)d_in[15];

    // ---- workspace carve-up ----
    char* ws = (char*)d_ws;
    size_t off = 0;
    auto alloc = [&](size_t bytes) -> void* {
        void* p = ws + off; off += (bytes + 255) & ~(size_t)255; return p;
    };
    float*  h1   = (float*)alloc((size_t)NB * 32  * HW * 4);
    float*  c1   = (float*)alloc((size_t)NB * 32  * HW * 4);
    float*  h2   = (float*)alloc((size_t)NB * 64  * HW * 4);
    float*  c2   = (float*)alloc((size_t)NB * 64  * HW * 4);
    float*  h3   = (float*)alloc((size_t)NB * 128 * HW * 4);
    float*  c3   = (float*)alloc((size_t)NB * 128 * HW * 4);
    float*  Z    = (float*)alloc((size_t)512 * NCOL * 4);
    __bf16* col  = (__bf16*)alloc((size_t)NCOL * 512 * 2);
    __bf16* wx1b = (__bf16*)alloc((size_t)128 * 32  * 2);
    __bf16* wh1b = (__bf16*)alloc((size_t)128 * 128 * 2);
    __bf16* wx2b = (__bf16*)alloc((size_t)256 * 128 * 2);
    __bf16* wh2b = (__bf16*)alloc((size_t)256 * 256 * 2);
    __bf16* wx3b = (__bf16*)alloc((size_t)512 * 256 * 2);
    __bf16* wh3b = (__bf16*)alloc((size_t)512 * 512 * 2);
    __bf16* w4t  = (__bf16*)alloc((size_t)256 * 131072 * 2);
    __bf16* w5t  = (__bf16*)alloc((size_t)1024 * 256 * 2);
    __bf16* fbuf = (__bf16*)alloc((size_t)NB * 131072 * 2);
    float*  Z4   = (float*)alloc((size_t)256 * NB * 4);
    __bf16* a4b  = (__bf16*)alloc((size_t)NB * 256 * 2);
    float*  Z5   = (float*)alloc((size_t)1024 * NB * 4);

    auto gemm = [&](const __bf16* A, const __bf16* B, float* C, const float* bias,
                    int M, int N, int K, int acc, int relu) {
        dim3 grid(cdiv((size_t)N, 512), M / 32);
        gemm_wmma_bf16<<<grid, 256, 0, stream>>>(A, B, C, bias, M, N, K, acc, relu);
    };
    auto im2row = [&](const float* src, int C, int Kpad, int bstride, int soff) {
        size_t total = (size_t)NCOL * Kpad;
        im2row_kernel<<<cdiv(total, 256), 256, 0, stream>>>(src, col, C, Kpad, bstride, soff);
    };
    auto gates = [&](float* h, float* c, int F) {
        size_t total = (size_t)NB * F * HW;
        lstm_gates_kernel<<<cdiv(total, 256), 256, 0, stream>>>(Z, h, c, F);
    };
    auto zero = [&](float* p, size_t n) {
        zero_kernel<<<cdiv(n, 256), 256, 0, stream>>>(p, n);
    };

    // ---- one-time weight conversion (per call; deterministic) ----
    cvt_w_kernel<<<cdiv((size_t)128 * 32,  256), 256, 0, stream>>>(Wx1, wx1b, 128, 4,   32);
    cvt_w_kernel<<<cdiv((size_t)128 * 128, 256), 256, 0, stream>>>(Wh1, wh1b, 128, 128, 128);
    cvt_w_kernel<<<cdiv((size_t)256 * 128, 256), 256, 0, stream>>>(Wx2, wx2b, 256, 128, 128);
    cvt_w_kernel<<<cdiv((size_t)256 * 256, 256), 256, 0, stream>>>(Wh2, wh2b, 256, 256, 256);
    cvt_w_kernel<<<cdiv((size_t)512 * 256, 256), 256, 0, stream>>>(Wx3, wx3b, 512, 256, 256);
    cvt_w_kernel<<<cdiv((size_t)512 * 512, 256), 256, 0, stream>>>(Wh3, wh3b, 512, 512, 512);
    cvt_t_kernel<<<cdiv((size_t)131072 * 256, 256), 256, 0, stream>>>(W4, w4t, 131072, 256);
    cvt_t_kernel<<<cdiv((size_t)256 * 1024,   256), 256, 0, stream>>>(W5, w5t, 256, 1024);

    zero(h1, (size_t)NB * 32  * HW); zero(c1, (size_t)NB * 32  * HW);
    zero(h2, (size_t)NB * 64  * HW); zero(c2, (size_t)NB * 64  * HW);
    zero(h3, (size_t)NB * 128 * HW); zero(c3, (size_t)NB * 128 * HW);

    // ---- recurrence: per-timestep, layers interleaved ----
    for (int t = 0; t < NSTEP; ++t) {
        // layer 1 (F=32): x conv (K=4 padded to 32) + h conv (K=128)
        im2row(x, 1, 32, NSTEP * HW, t * HW);
        gemm(wx1b, col, Z, b1, 128, NCOL, 32, 0, 0);
        im2row(h1, 32, 128, 32 * HW, 0);
        gemm(wh1b, col, Z, nullptr, 128, NCOL, 128, 1, 0);
        gates(h1, c1, 32);

        // layer 2 (F=64): x = h1
        im2row(h1, 32, 128, 32 * HW, 0);
        gemm(wx2b, col, Z, b2, 256, NCOL, 128, 0, 0);
        im2row(h2, 64, 256, 64 * HW, 0);
        gemm(wh2b, col, Z, nullptr, 256, NCOL, 256, 1, 0);
        gates(h2, c2, 64);

        // layer 3 (F=128): x = h2
        im2row(h2, 64, 256, 64 * HW, 0);
        gemm(wx3b, col, Z, b3, 512, NCOL, 256, 0, 0);
        im2row(h3, 128, 512, 128 * HW, 0);
        gemm(wh3b, col, Z, nullptr, 512, NCOL, 512, 1, 0);
        gates(h3, c3, 128);
    }

    // ---- dense head (channel-major activations) ----
    cvt_flat_kernel<<<cdiv((size_t)NB * 131072, 256), 256, 0, stream>>>(
        h3, fbuf, (size_t)NB * 131072);
    gemm(w4t, fbuf, Z4, b4, 256, NB, 131072, 0, 1);   // a4^T = relu(W4^T f + b4)
    cvt_t_kernel<<<cdiv((size_t)256 * NB, 256), 256, 0, stream>>>(Z4, a4b, 256, NB);
    gemm(w5t, a4b, Z5, b5, 1024, NB, 256, 0, 1);      // a5^T
    head_kernel<<<1, 64, 0, stream>>>(Z5, W6, b6, (float*)d_out);
}
